// RNNLayer_9320079033009
// MI455X (gfx1250) — matrix operand order repaired and checked by reading, not compiled, not run on hardware
//
#include <hip/hip_runtime.h>

// ---------------------------------------------------------------------------
// LSTM layer for MI455X (gfx1250, wave32, WMMA bf16 + TDM).
// T=512, B=32, I=512, H=1024.
//   1) convert x, W_ih, W_hh to bf16 in workspace
//   2) massively parallel WMMA GEMM: xg = x @ W_ih^T + (b_ih+b_hh)  (fp32)
//   3) persistent cooperative kernel (16 blocks x 256 thr = 128 waves) doing
//      all 512 recurrent steps.  Per step, each block DMAs h [32x1024 bf16,
//      64KB] into LDS with TENSOR_LOAD_TO_LDS (TENSORcnt), reads A-fragments
//      from LDS, streams W_hh B-fragments from L2 (8.4MB bf16, L2-resident),
//      and does the LSTM pointwise math in-register on WMMA accumulators.
//      Steps are separated by an L2 atomic grid barrier (per-t counters).
// ---------------------------------------------------------------------------

typedef __attribute__((ext_vector_type(16))) __bf16 v16bf;
typedef __attribute__((ext_vector_type(8)))  __bf16 v8bf;
typedef __attribute__((ext_vector_type(8)))  float  v8f;
typedef unsigned u32x4 __attribute__((ext_vector_type(4)));
typedef int      i32x8 __attribute__((ext_vector_type(8)));
typedef int      i32x4 __attribute__((ext_vector_type(4)));

#define T_STEPS 512
#define BATCH   32
#define I_DIM   512
#define H_DIM   1024
#define G_DIM   4096   // 4*H

#if __has_builtin(__builtin_amdgcn_tensor_load_to_lds) && \
    __has_builtin(__builtin_amdgcn_s_wait_tensorcnt)
#define USE_TDM 1
#else
#define USE_TDM 0
#endif

// workspace layout (bytes)
static constexpr size_t OFF_XBF  = 0;                                         // 16,777,216
static constexpr size_t OFF_WIH  = OFF_XBF + (size_t)T_STEPS*BATCH*I_DIM*2;   //  4,194,304
static constexpr size_t OFF_WHH  = OFF_WIH + (size_t)G_DIM*I_DIM*2;           //  8,388,608
static constexpr size_t OFF_BIAS = OFF_WHH + (size_t)G_DIM*H_DIM*2;           //     16,384
static constexpr size_t OFF_HBUF = OFF_BIAS + (size_t)G_DIM*4;                //  2x 65,536
static constexpr size_t OFF_C    = OFF_HBUF + 2*(size_t)BATCH*H_DIM*2;        //    131,072
static constexpr size_t OFF_CNT  = OFF_C + (size_t)BATCH*H_DIM*4;             //      2,048
static constexpr size_t OFF_XG   = (OFF_CNT + (size_t)T_STEPS*4 + 255) & ~(size_t)255; // 268,435,456
// total ~298 MB

__device__ __forceinline__ float sigmoidf_(float x) {
    return 1.0f / (1.0f + __expf(-x));
}

// A fragment: 16-bit A 16x32 (MxK).  lane<16: row=lane, K={0..7,16..23};
// lane>=16: row=lane-16, K={8..15,24..31}.  Two contiguous 16B loads.
__device__ __forceinline__ v16bf load_a_frag(const __bf16* A, int row, int ldk,
                                             int kk, int lane) {
    const __bf16* p = A + (size_t)row * ldk + kk * 32 + ((lane < 16) ? 0 : 8);
    v8bf lo = *(const v8bf*)p;
    v8bf hi = *(const v8bf*)(p + 16);
    v16bf a;
#pragma unroll
    for (int i = 0; i < 8; ++i) { a[i] = lo[i]; a[8 + i] = hi[i]; }
    return a;
}

// B fragment: 16-bit B 32x16 (KxN), B[k,n] = W[n,k] (W row-major [N,K]).
// lane<16: col=lane&15, K=kk*32+{0..7,16..23}; lane>=16: K=kk*32+{8..15,24..31}.
__device__ __forceinline__ v16bf load_b_frag(const __bf16* W, int col, int ldk,
                                             int kk, int lane) {
    const __bf16* p = W + (size_t)col * ldk + kk * 32 + ((lane < 16) ? 0 : 8);
    v8bf lo = *(const v8bf*)p;
    v8bf hi = *(const v8bf*)(p + 16);
    v16bf b;
#pragma unroll
    for (int i = 0; i < 8; ++i) { b[i] = lo[i]; b[8 + i] = hi[i]; }
    return b;
}

// ---------------------------------------------------------------- converts
__global__ void __launch_bounds__(256) k_cvt_bf16(const float* __restrict__ src,
                                                  __bf16* __restrict__ dst, int n) {
    int i = blockIdx.x * 256 + threadIdx.x;
    if (i < n) dst[i] = (__bf16)src[i];
}

__global__ void __launch_bounds__(256) k_init(const float* __restrict__ bih,
                                              const float* __restrict__ bhh,
                                              float* __restrict__ bias,
                                              const float* __restrict__ h0,
                                              const float* __restrict__ c0,
                                              __bf16* __restrict__ hbuf0,
                                              float* __restrict__ c,
                                              unsigned* __restrict__ cnt) {
    int i = blockIdx.x * 256 + threadIdx.x;           // 0 .. 32767
    if (i < G_DIM) bias[i] = bih[i] + bhh[i];
    if (i < BATCH * H_DIM) { hbuf0[i] = (__bf16)h0[i]; c[i] = c0[i]; }
    if (i < T_STEPS) cnt[i] = 0u;
}

// -------------------------------------------------- phase 1: xg = x@Wih^T+b
// M = T*B = 16384 (1024 tiles), N = 4096 (256 tiles), K = 512.
__global__ void __launch_bounds__(256) k_xgemm(const __bf16* __restrict__ Xbf,
                                               const __bf16* __restrict__ Wih,
                                               const float* __restrict__ bias,
                                               float* __restrict__ xg) {
    const int lane = threadIdx.x & 31;
    const int wid  = blockIdx.x * 8 + (threadIdx.x >> 5);
    const int mtile = wid >> 8;      // 0..1023
    const int ntile = wid & 255;     // 0..255
    const int M0 = mtile * 16;
    const int col = ntile * 16 + (lane & 15);
    const int arow = M0 + (lane & 15);

    v8f acc = {};
#pragma unroll 4
    for (int kk = 0; kk < I_DIM / 32; ++kk) {
        v16bf a = load_a_frag(Xbf, arow, I_DIM, kk, lane);
        v16bf b = load_b_frag(Wih, col, I_DIM, kk, lane);
        acc = __builtin_amdgcn_wmma_f32_16x16x32_bf16(false, a, false, b,
                                                      (short)0, acc, false, false);
    }
    const float bv = bias[col];
    const int mbase = M0 + ((lane < 16) ? 0 : 8);
#pragma unroll
    for (int r = 0; r < 8; ++r)
        xg[(size_t)(mbase + r) * G_DIM + col] = acc[r] + bv;
}

// ------------------------------------- phase 2: persistent recurrent kernel
__global__ void __launch_bounds__(256, 1)
k_lstm(const float* __restrict__ xg, const __bf16* __restrict__ Whh,
       __bf16* __restrict__ hbuf, float* __restrict__ c,
       float* __restrict__ out, unsigned* __restrict__ cnt) {
    __shared__ __bf16 hsh[BATCH * H_DIM];   // 64 KB staged copy of h_t

    const int lane  = threadIdx.x & 31;
    const int wid   = blockIdx.x * 8 + (threadIdx.x >> 5);  // 0..127
    const int mtile = wid & 1;
    const int ngrp  = wid >> 1;                             // 0..63
    const int M0    = mtile * 16;
    const int nl    = ngrp * 16 + (lane & 15);  // hidden col owned for C/D
    const int mbase = M0 + ((lane < 16) ? 0 : 8);
    const int arow  = M0 + (lane & 15);         // A row for h fragment

    float* hT = out + (size_t)T_STEPS * BATCH * H_DIM;
    float* cT = hT + (size_t)BATCH * H_DIM;

    // LDS byte address of hsh (flat LDS addresses truncate to LDS offset)
    const unsigned lds_addr = (unsigned)(unsigned long long)(uintptr_t)&hsh[0];

#if USE_TDM
    // D# group1 (constant across steps): data_size=3 (8B units),
    // tensor_dim0 = tile_dim0 = stride0 = 256 (8B units/row = 2048B),
    // tensor_dim1 = tile_dim1 = 32 rows.  2-D tensor -> groups 2/3 zero.
    const i32x8 g1 = { 0x00030000,          // mask=0 | data_size=3<<16
                       0x01000000,          // tensor_dim0[15:0]=256 << 16
                       0x00200000,          // tensor_dim1[15:0]=32 << 16
                       0x01000000,          // tile_dim0=256 << 16
                       32,                  // tile_dim1=32
                       256,                 // tensor_dim0_stride[31:0]=256
                       0, 0 };
    const i32x4 gz4 = { 0, 0, 0, 0 };
    const i32x8 gz8 = { 0, 0, 0, 0, 0, 0, 0, 0 };
#endif

    for (int t = 0; t < T_STEPS; ++t) {
        const __bf16* hcur  = hbuf + (size_t)(t & 1) * BATCH * H_DIM;
        __bf16*       hnext = hbuf + (size_t)((t + 1) & 1) * BATCH * H_DIM;
        const float*  xgt   = xg + (size_t)t * BATCH * G_DIM;

        // ---- stage h_t into LDS ------------------------------------------
#if USE_TDM
        if ((threadIdx.x >> 5) == 0) {      // one wave per block issues TDM
            unsigned long long ga = (unsigned long long)(uintptr_t)hcur;
            u32x4 g0;
            g0[0] = 1u;                                        // count=1
            g0[1] = lds_addr;                                  // lds_addr
            g0[2] = (unsigned)ga;                              // global[31:0]
            g0[3] = (unsigned)((ga >> 32) & 0x1FFFFFFu) | (2u << 30); // +type=2
            __builtin_amdgcn_tensor_load_to_lds(g0, g1, gz4, gz4, gz8, 0);
            __builtin_amdgcn_s_wait_tensorcnt(0);
        }
#else
        for (int i = threadIdx.x; i < BATCH * H_DIM / 8; i += 256)
            ((v8bf*)hsh)[i] = ((const v8bf*)hcur)[i];
#endif
        __syncthreads();

        // ---- seed accumulators with precomputed input projections --------
        v8f acc[4];
#pragma unroll
        for (int g = 0; g < 4; ++g)
#pragma unroll
            for (int r = 0; r < 8; ++r)
                acc[g][r] = xgt[(size_t)(mbase + r) * G_DIM + g * H_DIM + nl];

        // ---- gates += h @ W_hh^T (A from LDS, B from L2) -----------------
        for (int kk = 0; kk < H_DIM / 32; ++kk) {
            v16bf a = load_a_frag(hsh, arow, H_DIM, kk, lane);
            if (kk + 1 < H_DIM / 32) {   // uniform branch: prefetch next slab
                __builtin_prefetch((const void*)(Whh + (size_t)nl * H_DIM +
                                                 (kk + 1) * 32), 0, 3);
            }
#pragma unroll
            for (int g = 0; g < 4; ++g) {
                v16bf b = load_b_frag(Whh, g * H_DIM + nl, H_DIM, kk, lane);
                acc[g] = __builtin_amdgcn_wmma_f32_16x16x32_bf16(
                    false, a, false, b, (short)0, acc[g], false, false);
            }
        }

        // ---- LSTM pointwise, in-register on the accumulator tile ---------
#pragma unroll
        for (int r = 0; r < 8; ++r) {
            const int m = mbase + r;
            const size_t ci = (size_t)m * H_DIM + nl;
            float iv = sigmoidf_(acc[0][r]);
            float fv = sigmoidf_(acc[1][r]);
            float gv = tanhf(acc[2][r]);
            float ov = sigmoidf_(acc[3][r]);
            float cv = fv * c[ci] + iv * gv;
            float hv = ov * tanhf(cv);
            c[ci] = cv;
            out[((size_t)t * BATCH + m) * H_DIM + nl] = hv;
            hnext[ci] = (__bf16)hv;
            if (t == T_STEPS - 1) { hT[ci] = hv; cT[ci] = cv; }
        }

        // ---- grid-wide barrier (per-t counter, 16 blocks) ----------------
        __syncthreads();
        if (threadIdx.x == 0) {
            __threadfence();
            atomicAdd(&cnt[t], 1u);
            while (__hip_atomic_load(&cnt[t], __ATOMIC_ACQUIRE,
                                     __HIP_MEMORY_SCOPE_AGENT) < 16u)
                __builtin_amdgcn_s_sleep(1);
        }
        __syncthreads();
    }
}

// ---------------------------------------------------------------------------
extern "C" void kernel_launch(void* const* d_in, const int* in_sizes, int n_in,
                              void* d_out, int out_size, void* d_ws, size_t ws_size,
                              hipStream_t stream) {
    const float* x   = (const float*)d_in[0];
    const float* h0  = (const float*)d_in[1];
    const float* c0  = (const float*)d_in[2];
    const float* Wih = (const float*)d_in[3];
    const float* Whh = (const float*)d_in[4];
    const float* bih = (const float*)d_in[5];
    const float* bhh = (const float*)d_in[6];
    float* out = (float*)d_out;

    char* ws = (char*)d_ws;
    __bf16*   xbf   = (__bf16*)(ws + OFF_XBF);
    __bf16*   wihbf = (__bf16*)(ws + OFF_WIH);
    __bf16*   whhbf = (__bf16*)(ws + OFF_WHH);
    float*    bias  = (float*)(ws + OFF_BIAS);
    __bf16*   hbuf  = (__bf16*)(ws + OFF_HBUF);
    float*    cbuf  = (float*)(ws + OFF_C);
    unsigned* cnt   = (unsigned*)(ws + OFF_CNT);
    float*    xgbuf = (float*)(ws + OFF_XG);

    const int nx = T_STEPS * BATCH * I_DIM;   // 8,388,608
    const int ni = G_DIM * I_DIM;             // 2,097,152
    const int nh = G_DIM * H_DIM;             // 4,194,304

    k_cvt_bf16<<<(nx + 255) / 256, 256, 0, stream>>>(x,   xbf,   nx);
    k_cvt_bf16<<<(ni + 255) / 256, 256, 0, stream>>>(Wih, wihbf, ni);
    k_cvt_bf16<<<(nh + 255) / 256, 256, 0, stream>>>(Whh, whhbf, nh);
    k_init<<<128, 256, 0, stream>>>(bih, bhh, bias, h0, c0, hbuf, cbuf, cnt);

    // xg GEMM: 1024 * 256 tiles, 8 waves/block -> 32768 blocks
    k_xgemm<<<32768, 256, 0, stream>>>(xbf, wihbf, bias, xgbuf);

    // persistent recurrent kernel: exactly 16 cooperating blocks
    k_lstm<<<16, 256, 0, stream>>>(xgbuf, whhbf, hbuf, cbuf, out, cnt);
}
